// FFJORD_27900107554844
// MI455X (gfx1250) — matrix operand order repaired and checked by reading, not compiled, not run on hardware
//
#include <hip/hip_runtime.h>

// ---------------------------------------------------------------------------
// FFJORD on MI455X (gfx1250): 256 sequential 3-layer MLP evals over B=8192.
// Compute-bound (≈5.5 TFLOP vs ≈16 GB of unavoidable activation traffic at
// 23.3 TB/s) -> drive the WMMA pipe with bf16 inputs / f32 accumulation.
// Tile staging: double-buffered async global->LDS copies (ASYNCcnt) with
// partial s_wait_asynccnt so next-tile fetch overlaps current-tile WMMAs.
// ---------------------------------------------------------------------------

typedef __attribute__((ext_vector_type(16))) __bf16 v16bf;
typedef __attribute__((ext_vector_type(8)))  float  v8f;

union Frag { v16bf v; uint4 q[2]; };

__device__ __forceinline__ unsigned short f2bf(float f) {
    unsigned int u = __builtin_bit_cast(unsigned int, f);
    u += 0x7FFFu + ((u >> 16) & 1u);       // round-to-nearest-even
    return (unsigned short)(u >> 16);
}

// Async 16B global->LDS copy. For a generic pointer into LDS, addr[31:0] is
// the LDS byte address (flat aperture rule, ISA §10.2), which is exactly what
// the VDST operand of global_load_async_to_lds_* expects.
__device__ __forceinline__ void async_copy16(const void* gptr, void* lptr) {
    unsigned lds_addr = (unsigned)(unsigned long long)lptr;
    asm volatile("global_load_async_to_lds_b128 %0, %1, off"
                 :
                 : "v"(lds_addr), "v"(gptr)
                 : "memory");
}

#define TM 128
#define TN 128
#define TK 32

// Issue the 4 async copies (per wave) staging one 128x32 A tile and one
// 128x32 Bt tile into the given LDS buffers. 512 16B chunks per tile,
// 2 per thread.
__device__ __forceinline__ void stage_tiles(const unsigned short* __restrict__ A,
                                            const unsigned short* __restrict__ Bt,
                                            unsigned short* la, unsigned short* lb,
                                            int m0, int n0, int K, int k0, int tid)
{
    #pragma unroll
    for (int it = 0; it < 2; ++it) {
        int c   = tid + it * 256;
        int row = c >> 2;              // 32 bf16 per row = 4 chunks
        int col = (c & 3) * 8;
        async_copy16(&A[(size_t)(m0 + row) * K + k0 + col], &la[row * TK + col]);
        async_copy16(&Bt[(size_t)(n0 + row) * K + k0 + col], &lb[row * TK + col]);
    }
}

// C(MxN) = act(A(MxK,bf16) * Bt(NxK,bf16)^T + bias [+ t*wlast])
// Block: 128x128 tile, 256 threads = 8 waves (wave32) in 4(M) x 2(N) grid.
// Wave tile: 32x64 = 2x4 WMMA 16x16 accumulators, K stepped by 32.
template<bool TANH, bool HAS_T>
__global__ __launch_bounds__(256)
void gemm_bias_act(const unsigned short* __restrict__ A,
                   const unsigned short* __restrict__ Bt,
                   const float* __restrict__ bias,
                   const float* __restrict__ wlast,
                   float tval,
                   float* __restrict__ outF,
                   unsigned short* __restrict__ outBf,
                   int M, int N, int K)
{
    __shared__ unsigned short lsA[2][TM * TK];
    __shared__ unsigned short lsB[2][TN * TK];

    const int tid   = threadIdx.x;
    const int lane  = tid & 31;
    const int wave  = tid >> 5;
    const int waveM = wave >> 1;   // 0..3
    const int waveN = wave & 1;    // 0..1
    const int m0 = blockIdx.y * TM;
    const int n0 = blockIdx.x * TN;

    v8f acc[2][4] = {};

    // Prologue: stage first K-chunk into buffer 0.
    stage_tiles(A, Bt, lsA[0], lsB[0], m0, n0, K, 0, tid);

    int p = 0;
    for (int k0 = 0; k0 < K; k0 += TK, p ^= 1) {
        const bool has_next = (k0 + TK) < K;

        // Kick off next tile into the other buffer, then wait only for the
        // OLDER group of 4 async copies (in-order completion, §7.1): current
        // buffer is ready while the next tile streams in during the WMMAs.
        if (has_next) {
            stage_tiles(A, Bt, lsA[p ^ 1], lsB[p ^ 1], m0, n0, K, k0 + TK, tid);
            asm volatile("s_wait_asynccnt 0x4" ::: "memory");
        } else {
            asm volatile("s_wait_asynccnt 0x0" ::: "memory");
        }
        __syncthreads();

        const unsigned short* la = lsA[p];
        const unsigned short* lb = lsB[p];

        // A fragment (ISA 7.12.2, 16-bit A 16x32): lane row = lane&15,
        // VGPR0-3 = K {0..7 | 8..15 per half-wave}, VGPR4-7 = same +16.
        Frag a[2];
        #pragma unroll
        for (int mt = 0; mt < 2; ++mt) {
            int row = waveM * 32 + mt * 16 + (lane & 15);
            int kb  = (lane >> 4) * 8;
            a[mt].q[0] = *(const uint4*)&la[row * TK + kb];
            a[mt].q[1] = *(const uint4*)&la[row * TK + kb + 16];
        }
        // B fragment (32x16): lane col = lane&15, same K order as A:
        // VGPR0-3 = K {0..7 | 8..15 per half-wave}, VGPR4-7 = same +16.
        Frag b[4];
        #pragma unroll
        for (int nt = 0; nt < 4; ++nt) {
            int row = waveN * 64 + nt * 16 + (lane & 15);
            int kb  = (lane >> 4) * 8;
            b[nt].q[0] = *(const uint4*)&lb[row * TK + kb];
            b[nt].q[1] = *(const uint4*)&lb[row * TK + kb + 16];
        }

        #pragma unroll
        for (int mt = 0; mt < 2; ++mt)
            #pragma unroll
            for (int nt = 0; nt < 4; ++nt)
                acc[mt][nt] = __builtin_amdgcn_wmma_f32_16x16x32_bf16(
                    false, a[mt].v, false, b[nt].v,
                    (short)0, acc[mt][nt], false, false);

        // Keep next iteration's async writes from racing readers of buf p.
        __syncthreads();
    }

    // Epilogue: bias (+ t*W1_last_row), tanh, fused fp32 / bf16 stores.
    #pragma unroll
    for (int nt = 0; nt < 4; ++nt) {
        int col = n0 + waveN * 64 + nt * 16 + (lane & 15);
        float bv = bias[col];
        if (HAS_T) bv += tval * wlast[col];
        #pragma unroll
        for (int mt = 0; mt < 2; ++mt) {
            int rbase = m0 + waveM * 32 + mt * 16 + ((lane >> 4) * 8);
            #pragma unroll
            for (int r = 0; r < 8; ++r) {
                float v = acc[mt][nt][r] + bv;
                if (TANH) v = tanhf(v);
                size_t idx = (size_t)(rbase + r) * N + col;
                if (outF)  outF[idx]  = v;
                if (outBf) outBf[idx] = f2bf(v);
            }
        }
    }
}

// Wt[n*K + k] = bf16(W[k*N + n])  (one-time weight transpose+convert)
__global__ void conv_tr(const float* __restrict__ W,
                        unsigned short* __restrict__ Wt, int K, int N)
{
    int idx = blockIdx.x * blockDim.x + threadIdx.x;
    if (idx >= K * N) return;
    int k = idx / N, n = idx % N;
    Wt[(size_t)n * K + k] = f2bf(W[idx]);
}

__global__ void init_kernel(const float* __restrict__ in,
                            float* __restrict__ x,
                            unsigned short* __restrict__ xbf, int n)
{
    int i = blockIdx.x * blockDim.x + threadIdx.x;
    if (i >= n) return;
    float v = in[i];
    x[i]   = v;
    xbf[i] = f2bf(v);
}

// RK4 stage combine: ksum accumulation + next stage input (bf16).
__global__ void stage_kernel(float* __restrict__ x,
                             const float* __restrict__ kbuf,
                             float* __restrict__ ksum,
                             unsigned short* __restrict__ xbf,
                             float kc, float a, float dt6,
                             int init, int finalize, int n)
{
    int i = blockIdx.x * blockDim.x + threadIdx.x;
    if (i >= n) return;
    float k = kbuf[i];
    float s = init ? k : (ksum[i] + kc * k);
    ksum[i] = s;
    float xv = finalize ? (x[i] + dt6 * s) : (x[i] + a * k);
    if (finalize) x[i] = xv;
    xbf[i] = f2bf(xv);
}

__global__ void copy_kernel(const float* __restrict__ src,
                            float* __restrict__ dst, int n)
{
    int i = blockIdx.x * blockDim.x + threadIdx.x;
    if (i < n) dst[i] = src[i];
}

extern "C" void kernel_launch(void* const* d_in, const int* in_sizes, int n_in,
                              void* d_out, int out_size, void* d_ws, size_t ws_size,
                              hipStream_t stream)
{
    (void)in_sizes; (void)n_in; (void)out_size; (void)ws_size;

    const int B = 8192, D = 128, H = 1024, NB = 2, NS = 32;
    const float dt = 1.0f / NS;

    const float* inx = (const float*)d_in[0];
    const float* W1  = (const float*)d_in[1];   // NB x (D+1) x H
    const float* b1  = (const float*)d_in[2];   // NB x H
    const float* W2  = (const float*)d_in[3];   // NB x H x H
    const float* b2  = (const float*)d_in[4];   // NB x H
    const float* W3  = (const float*)d_in[5];   // NB x H x D
    const float* b3  = (const float*)d_in[6];   // NB x D

    char* ws = (char*)d_ws;
    size_t off = 0;
    auto alloc = [&](size_t bytes) -> char* {
        char* p = ws + off;
        off = (off + bytes + 255) & ~(size_t)255;
        return p;
    };
    unsigned short* Wt1 = (unsigned short*)alloc((size_t)NB * H * D * 2);
    unsigned short* Wt2 = (unsigned short*)alloc((size_t)NB * H * H * 2);
    unsigned short* Wt3 = (unsigned short*)alloc((size_t)NB * D * H * 2);
    float*          x    = (float*)alloc((size_t)B * D * 4);
    float*          ksum = (float*)alloc((size_t)B * D * 4);
    float*          kbuf = (float*)alloc((size_t)B * D * 4);
    unsigned short* xbf  = (unsigned short*)alloc((size_t)B * D * 2);
    unsigned short* h1bf = (unsigned short*)alloc((size_t)B * H * 2);
    unsigned short* h2bf = (unsigned short*)alloc((size_t)B * H * 2);

    // One-time (per launch) weight transpose + bf16 convert.
    for (int i = 0; i < NB; ++i) {
        conv_tr<<<dim3((D * H + 255) / 256), 256, 0, stream>>>(
            W1 + (size_t)i * (D + 1) * H, Wt1 + (size_t)i * H * D, D, H);
        conv_tr<<<dim3((H * H + 255) / 256), 256, 0, stream>>>(
            W2 + (size_t)i * H * H, Wt2 + (size_t)i * H * H, H, H);
        conv_tr<<<dim3((H * D + 255) / 256), 256, 0, stream>>>(
            W3 + (size_t)i * H * D, Wt3 + (size_t)i * D * H, H, D);
    }
    init_kernel<<<(B * D + 255) / 256, 256, 0, stream>>>(inx, x, xbf, B * D);

    dim3 blk(256);
    dim3 g1(H / TN, B / TM);   // 8 x 64
    dim3 g2(H / TN, B / TM);   // 8 x 64
    dim3 g3(D / TN, B / TM);   // 1 x 64

    for (int i = 0; i < NB; ++i) {
        const unsigned short* wt1 = Wt1 + (size_t)i * H * D;
        const unsigned short* wt2 = Wt2 + (size_t)i * H * H;
        const unsigned short* wt3 = Wt3 + (size_t)i * D * H;
        const float* c1 = b1 + (size_t)i * H;
        const float* c2 = b2 + (size_t)i * H;
        const float* c3 = b3 + (size_t)i * D;
        const float* w1last = W1 + (size_t)i * (D + 1) * H + (size_t)D * H;

        for (int s = 0; s < NS; ++s) {
            float t0 = s * dt;
            const float ts[4] = { t0, t0 + 0.5f * dt, t0 + 0.5f * dt, t0 + dt };
            for (int st = 0; st < 4; ++st) {
                gemm_bias_act<true, true><<<g1, blk, 0, stream>>>(
                    xbf, wt1, c1, w1last, ts[st], nullptr, h1bf, B, H, D);
                gemm_bias_act<true, false><<<g2, blk, 0, stream>>>(
                    h1bf, wt2, c2, nullptr, 0.f, nullptr, h2bf, B, H, H);
                gemm_bias_act<false, false><<<g3, blk, 0, stream>>>(
                    h2bf, wt3, c3, nullptr, 0.f, kbuf, nullptr, B, D, H);

                float kc = (st == 1 || st == 2) ? 2.f : 1.f;
                float a  = (st == 0 || st == 1) ? 0.5f * dt : dt;
                stage_kernel<<<(B * D + 255) / 256, 256, 0, stream>>>(
                    x, kbuf, ksum, xbf, kc, a, dt / 6.f,
                    st == 0 ? 1 : 0, st == 3 ? 1 : 0, B * D);
            }
        }
    }
    copy_kernel<<<(B * D + 255) / 256, 256, 0, stream>>>(x, (float*)d_out, B * D);
}
